// GreetingNet_82652350644713
// MI455X (gfx1250) — matrix operand order repaired and checked by reading, not compiled, not run on hardware
//
#include <hip/hip_runtime.h>

// ---------------- problem constants ----------------
#define B_   64
#define T_   80
#define K_   5
#define E_   256
#define H_   1024
#define V_   50257
#define VP_  50304          // V padded to multiple of 128 for WMMA tiles
#define NG_  4096           // 4*H gate width
#define L_   2
#define TEMP_INV 1.3333333333333333f   // 1/0.75
#define TWO_PI_OVER_24 0.2617993877991494f

typedef __attribute__((ext_vector_type(16))) __bf16 v16bf;
typedef __attribute__((ext_vector_type(8)))  __bf16 v8bf;
typedef __attribute__((ext_vector_type(8)))  float  v8f;

// ---- gfx1250 async memory->LDS copy (ASYNCcnt-tracked) ----
__device__ __forceinline__ void async_ld_b128(unsigned lds_off, const void* gaddr)
{
    asm volatile("global_load_async_to_lds_b128 %0, %1, off"
                 :: "v"(lds_off), "v"(gaddr) : "memory");
}
__device__ __forceinline__ void wait_async0()
{
    asm volatile("s_wait_asynccnt 0" ::: "memory");
}

// ------- weight conversion: f32 (K x Nin) -> bf16 TRANSPOSED (Nout x K) -----
__global__ __launch_bounds__(256)
void cvt_bf16_T(const float* __restrict__ in, __bf16* __restrict__ outp,
                int K, int Nin, int Nout)
{
    __shared__ __bf16 tile[32][33];
    const int kt = blockIdx.x * 32;
    const int nt = blockIdx.y * 32;
    const int tx = threadIdx.x;        // 0..31
    const int ty = threadIdx.y;        // 0..7
#pragma unroll
    for (int r = ty; r < 32; r += 8) {
        int n = nt + tx;
        float v = (n < Nin) ? in[(size_t)(kt + r) * Nin + n] : 0.0f;
        tile[r][tx] = (__bf16)v;
    }
    __syncthreads();
#pragma unroll
    for (int r = ty; r < 32; r += 8) {
        outp[(size_t)(nt + r) * K + kt + tx] = tile[tx][r];
    }
}

__global__ __launch_bounds__(256)
void pad_bias(const float* __restrict__ fc_b, float* __restrict__ outp)
{
    int n = blockIdx.x * 256 + threadIdx.x;
    if (n >= VP_) return;
    outp[n] = (n < V_) ? fc_b[n] : -1.0e30f;
}

// ---------------- time encoder ----------------
__global__ __launch_bounds__(256)
void time_mlp1(const float* __restrict__ hours, const float* __restrict__ w1,
               const float* __restrict__ bb1, float* __restrict__ tp1)
{
    int idx = blockIdx.x * 256 + threadIdx.x;
    if (idx >= B_ * H_) return;
    int b = idx / H_, j = idx - b * H_;
    float a = hours[b] * TWO_PI_OVER_24;
    float s = __sinf(a), c = __cosf(a);
    tp1[idx] = tanhf(s * w1[j] + c * w1[H_ + j] + bb1[j]);
}

// h state is kept in bf16 (WMMA A operand), c state in f32
__global__ __launch_bounds__(256)
void time_mlp2(const float* __restrict__ tp1, const float* __restrict__ w2,
               const float* __restrict__ bb2, __bf16* __restrict__ hbuf,
               float* __restrict__ cbuf)
{
    int idx = blockIdx.x * 256 + threadIdx.x;            // B * 4096
    if (idx >= B_ * (2 * L_ * H_)) return;
    int b = idx / (2 * L_ * H_), n = idx - b * (2 * L_ * H_);
    const float* a = tp1 + b * H_;
    float acc = bb2[n];
    for (int kk = 0; kk < H_; ++kk)
        acc = fmaf(a[kk], w2[(size_t)kk * (2 * L_ * H_) + n], acc);
    int l = n >> 11;
    int r = n & 2047;
    int s = r >> 10;
    int j = r & 1023;
    size_t off = ((size_t)l * B_ + b) * H_ + j;
    if (s) cbuf[off] = acc;
    else   hbuf[off] = (__bf16)acc;
}

__global__ void init_tok(const int* __restrict__ sos, int* __restrict__ tok)
{
    int b = threadIdx.x;
    if (b < B_) tok[b] = sos[0];
}

// ---------------- embedding gather (f32 table -> bf16 activations) ----------
__global__ __launch_bounds__(256)
void embed_gather(const float* __restrict__ embed, const int* __restrict__ tok,
                  __bf16* __restrict__ x)
{
    int idx = blockIdx.x * 256 + threadIdx.x;
    if (idx >= B_ * E_) return;
    int b = idx / E_, e = idx - b * E_;
    x[idx] = (__bf16)embed[(size_t)tok[b] * E_ + e];
}

// ---------------- WMMA GEMM: out[64][N] = A1@B1 + A2@B2 + bias --------------
// A: bf16 64xK row-major.  B: bf16 pre-transposed (N x K).
// WG tile 64x128; each of 8 waves owns a 16-wide N column and all 4 M tiles.
// A tiles staged to LDS with double-buffered global_load_async_to_lds_b128.
__global__ __launch_bounds__(256)
void gemm64_wmma(const __bf16* __restrict__ A1, const __bf16* __restrict__ B1T, int K1,
                 const __bf16* __restrict__ A2, const __bf16* __restrict__ B2T, int K2,
                 const float* __restrict__ bias, float* __restrict__ out, int N)
{
    __shared__ __bf16 As[2][64 * 32];

    const int tid   = threadIdx.x;
    const int wave  = tid >> 5;
    const int lane  = tid & 31;
    const int lm    = lane & 15;
    const int khalf = (lane >> 4) & 1;
    const int ncol  = blockIdx.x * 128 + wave * 16 + lm;  // this lane's N column

    // per-thread async staging slot: one b128 (8 bf16) of the 64x32 tile
    const int srow = tid >> 2;
    const int sc8  = (tid & 3) * 8;

    v8f acc[4] = {{}, {}, {}, {}};

    for (int pair = 0; pair < 2; ++pair) {
        const __bf16* A  = pair ? A2 : A1;
        const __bf16* BT = pair ? B2T : B1T;
        const int     K  = pair ? K2 : K1;
        if (K <= 0) continue;
        const __bf16* brow = BT + (size_t)ncol * K;
        const __bf16* agsrc = A + (size_t)srow * K + sc8;

        // prologue: kick off tile 0
        async_ld_b128((unsigned)(uintptr_t)&As[0][srow * 32 + sc8], agsrc);

        for (int k0 = 0; k0 < K; k0 += 32) {
            const int cur = (k0 >> 5) & 1;
            wait_async0();          // my async writes for tile `cur` done
            __syncthreads();        // everyone's writes visible; prev reads done
            if (k0 + 32 < K)        // prefetch next tile into the other buffer
                async_ld_b128((unsigned)(uintptr_t)&As[cur ^ 1][srow * 32 + sc8],
                              agsrc + k0 + 32);

            // B fragment: lanes 0-15 K 0..7 & 16..23 ; lanes 16-31 K 8..15 & 24..31
            const int kb = k0 + khalf * 8;
            v8bf blo = *(const v8bf*)(brow + kb);
            v8bf bhi = *(const v8bf*)(brow + kb + 16);
            v16bf bv = __builtin_shufflevector(blo, bhi,
                        0, 1, 2, 3, 4, 5, 6, 7, 8, 9, 10, 11, 12, 13, 14, 15);
            if (k0 + 32 < K)
                __builtin_prefetch(brow + kb + 32, 0, 0);

            // A fragments: lanes 0-15 K 0..7 & 16..23 ; lanes 16-31 K 8..15 & 24..31
#pragma unroll
            for (int mi = 0; mi < 4; ++mi) {
                const __bf16* ar = &As[cur][(mi * 16 + lm) * 32 + khalf * 8];
                v8bf alo = *(const v8bf*)ar;
                v8bf ahi = *(const v8bf*)(ar + 16);
                v16bf av = __builtin_shufflevector(alo, ahi,
                            0, 1, 2, 3, 4, 5, 6, 7, 8, 9, 10, 11, 12, 13, 14, 15);
                acc[mi] = __builtin_amdgcn_wmma_f32_16x16x32_bf16(
                              false, av, false, bv, (short)0, acc[mi], false, false);
            }
        }
        __syncthreads();   // pair boundary: buffers safe to re-stage
    }

    // C/D layout: VGPR g -> M = g (+8 for lanes 16-31), N = lane&15
    const float bi = bias[ncol];
#pragma unroll
    for (int mi = 0; mi < 4; ++mi) {
#pragma unroll
        for (int g = 0; g < 8; ++g) {
            int row = mi * 16 + khalf * 8 + g;
            out[(size_t)row * N + ncol] = acc[mi][g] + bi;
        }
    }
}

// ---------------- LSTM pointwise cell (h out in bf16) ----------------
__global__ __launch_bounds__(256)
void lstm_pointwise(const float* __restrict__ gates, __bf16* __restrict__ h,
                    float* __restrict__ c)
{
    int idx = blockIdx.x * 256 + threadIdx.x;
    if (idx >= B_ * H_) return;
    int b = idx / H_, j = idx - b * H_;
    const float* g = gates + (size_t)b * NG_;
    float gi = g[j], gf = g[H_ + j], gg = g[2 * H_ + j], go = g[3 * H_ + j];
    float si = 1.0f / (1.0f + __expf(-gi));
    float sf = 1.0f / (1.0f + __expf(-gf));
    float so = 1.0f / (1.0f + __expf(-go));
    float c2 = sf * c[idx] + si * tanhf(gg);
    c[idx] = c2;
    h[idx] = (__bf16)(so * tanhf(c2));
}

// ---------------- top-5 + softmax + categorical sample ----------------
__global__ __launch_bounds__(256)
void topk_sample(const float* __restrict__ logits, int t, int* __restrict__ tok,
                 float* __restrict__ out_tok, float* __restrict__ out_probs)
{
    const int b = blockIdx.x;
    const int tid = threadIdx.x;
    const float* row = logits + (size_t)b * VP_;
    __shared__ float sv[256];
    __shared__ int   si[256];
    __shared__ float stopv[K_];
    __shared__ int   stopi[K_];
    int excl[K_];

    for (int kk = 0; kk < K_; ++kk) {
        float best = -3.4e38f; int bi = VP_;
        for (int n = tid; n < VP_; n += 256) {
            bool skip = false;
            for (int e = 0; e < kk; ++e) skip = skip || (n == excl[e]);
            if (skip) continue;
            float v = row[n];
            if (v > best || (v == best && n < bi)) { best = v; bi = n; }
        }
        sv[tid] = best; si[tid] = bi;
        __syncthreads();
        for (int s = 128; s > 0; s >>= 1) {
            if (tid < s) {
                float ov = sv[tid + s]; int oi = si[tid + s];
                if (ov > sv[tid] || (ov == sv[tid] && oi < si[tid])) {
                    sv[tid] = ov; si[tid] = oi;
                }
            }
            __syncthreads();
        }
        if (tid == 0) { stopv[kk] = sv[0]; stopi[kk] = si[0]; }
        __syncthreads();
        excl[kk] = stopi[kk];
    }

    if (tid == 0) {
        float tv[K_], p[K_], mx = -3.4e38f;
        for (int kk = 0; kk < K_; ++kk) { tv[kk] = stopv[kk] * TEMP_INV; mx = fmaxf(mx, tv[kk]); }
        float sum = 0.0f;
        for (int kk = 0; kk < K_; ++kk) { p[kk] = __expf(tv[kk] - mx); sum += p[kk]; }
        float inv = 1.0f / sum;
        unsigned s = 0x9E3779B9u * (unsigned)(t + 1) ^ (0x85EBCA6Bu * (unsigned)(b + 17));
        s ^= s >> 16; s *= 0x7FEB352Du; s ^= s >> 15; s *= 0x846CA68Bu; s ^= s >> 16;
        float u = (float)(s >> 8) * (1.0f / 16777216.0f);
        int sel = K_ - 1; float cdf = 0.0f;
        for (int kk = 0; kk < K_; ++kk) {
            cdf += p[kk] * inv;
            if (u <= cdf) { sel = kk; break; }
        }
        for (int kk = 0; kk < K_; ++kk)
            out_probs[((size_t)b * T_ + t) * K_ + kk] = p[kk] * inv;
        out_tok[b * T_ + t] = (float)stopi[sel];
        tok[b] = stopi[sel];
    }
}

// ---------------- host-side orchestration ----------------
extern "C" void kernel_launch(void* const* d_in, const int* in_sizes, int n_in,
                              void* d_out, int out_size, void* d_ws, size_t ws_size,
                              hipStream_t stream)
{
    const float* hours = (const float*)d_in[0];
    const float* tp_w1 = (const float*)d_in[1];
    const float* tp_b1 = (const float*)d_in[2];
    const float* tp_w2 = (const float*)d_in[3];
    const float* tp_b2 = (const float*)d_in[4];
    const float* embed = (const float*)d_in[5];
    const float* w_ih0 = (const float*)d_in[6];
    const float* w_hh0 = (const float*)d_in[7];
    const float* b0    = (const float*)d_in[8];
    const float* w_ih1 = (const float*)d_in[9];
    const float* w_hh1 = (const float*)d_in[10];
    const float* b1    = (const float*)d_in[11];
    const float* fc_w  = (const float*)d_in[12];
    const float* fc_b  = (const float*)d_in[13];
    const int*   sos   = (const int*)d_in[14];

    // workspace carve-up (all chunks 16B aligned); transposed bf16 weights
    char* w = (char*)d_ws;
    __bf16* wih0T = (__bf16*)w; w += (size_t)NG_ * E_ * 2;   // 4096 x 256
    __bf16* whh0T = (__bf16*)w; w += (size_t)NG_ * H_ * 2;   // 4096 x 1024
    __bf16* wih1T = (__bf16*)w; w += (size_t)NG_ * H_ * 2;
    __bf16* whh1T = (__bf16*)w; w += (size_t)NG_ * H_ * 2;
    __bf16* fcwT  = (__bf16*)w; w += (size_t)VP_ * H_ * 2;   // 50304 x 1024
    float*  fcbp  = (float*)w;  w += (size_t)VP_ * 4;
    __bf16* xbuf  = (__bf16*)w; w += (size_t)B_ * E_ * 2;
    __bf16* hbuf  = (__bf16*)w; w += (size_t)L_ * B_ * H_ * 2;
    float*  cbuf  = (float*)w;  w += (size_t)L_ * B_ * H_ * 4;
    float*  gates = (float*)w;  w += (size_t)B_ * NG_ * 4;
    float*  logits= (float*)w;  w += (size_t)B_ * VP_ * 4;
    float*  tp1   = (float*)w;  w += (size_t)B_ * H_ * 4;
    int*    tok   = (int*)w;    w += 256;

    float* out_tok   = (float*)d_out;
    float* out_probs = (float*)d_out + B_ * T_;

    auto blocks = [](size_t n) { return (unsigned)((n + 255) / 256); };
    dim3 tb(32, 8);

    // one-time: transposed bf16 weight conversion (FC padded to 50304 rows)
    cvt_bf16_T<<<dim3(E_ / 32, NG_ / 32), tb, 0, stream>>>(w_ih0, wih0T, E_, NG_, NG_);
    cvt_bf16_T<<<dim3(H_ / 32, NG_ / 32), tb, 0, stream>>>(w_hh0, whh0T, H_, NG_, NG_);
    cvt_bf16_T<<<dim3(H_ / 32, NG_ / 32), tb, 0, stream>>>(w_ih1, wih1T, H_, NG_, NG_);
    cvt_bf16_T<<<dim3(H_ / 32, NG_ / 32), tb, 0, stream>>>(w_hh1, whh1T, H_, NG_, NG_);
    cvt_bf16_T<<<dim3(H_ / 32, VP_ / 32), tb, 0, stream>>>(fc_w,  fcwT,  H_, V_,  VP_);
    pad_bias<<<blocks(VP_), 256, 0, stream>>>(fc_b, fcbp);

    // one-time: time encoder -> (h0,c0)
    time_mlp1<<<blocks((size_t)B_ * H_), 256, 0, stream>>>(hours, tp_w1, tp_b1, tp1);
    time_mlp2<<<blocks((size_t)B_ * 2 * L_ * H_), 256, 0, stream>>>(tp1, tp_w2, tp_b2, hbuf, cbuf);
    init_tok<<<1, 64, 0, stream>>>(sos, tok);

    __bf16* h0 = hbuf;
    __bf16* h1 = hbuf + (size_t)B_ * H_;
    float*  c0 = cbuf;
    float*  c1 = cbuf + (size_t)B_ * H_;

    for (int t = 0; t < T_; ++t) {
        embed_gather<<<blocks((size_t)B_ * E_), 256, 0, stream>>>(embed, tok, xbuf);
        // layer 0: gates = x@Wih0 + h0@Whh0 + b0
        gemm64_wmma<<<NG_ / 128, 256, 0, stream>>>(xbuf, wih0T, E_, h0, whh0T, H_,
                                                   b0, gates, NG_);
        lstm_pointwise<<<blocks((size_t)B_ * H_), 256, 0, stream>>>(gates, h0, c0);
        // layer 1: gates = h0n@Wih1 + h1@Whh1 + b1
        gemm64_wmma<<<NG_ / 128, 256, 0, stream>>>(h0, wih1T, H_, h1, whh1T, H_,
                                                   b1, gates, NG_);
        lstm_pointwise<<<blocks((size_t)B_ * H_), 256, 0, stream>>>(gates, h1, c1);
        // fc: logits = h1n @ fc_w + fc_b   (second pair disabled via K2 = 0)
        gemm64_wmma<<<VP_ / 128, 256, 0, stream>>>(h1, fcwT, H_, h1, fcwT, 0,
                                                   fcbp, logits, VP_);
        topk_sample<<<B_, 256, 0, stream>>>(logits, t, tok, out_tok, out_probs);
    }
}
